// BestRQLayer_8375186227838
// MI455X (gfx1250) — matrix operand order repaired and checked by reading, not compiled, not run on hardware
//
#include <hip/hip_runtime.h>
#include <hip/hip_bf16.h>

// ---------------- problem constants ----------------
#define B_      32
#define T_      2048
#define D_      80
#define T2_     511          // ((2048-3)/2+1 - 3)/2 + 1
#define CBDIM   512
#define CBSIZE  1024
#define KIN     720          // D*K1*K2
#define KP      736          // KIN padded to 23*32
#define NKT1    23           // KP/32
#define MROWS   16352        // B*T2
#define MPAD    16384        // padded to 256*64
#define NFEAT   5242880      // B*T*D

// triple-buffered staging: per buffer 1024 dwords A + 4096 dwords B = 20KB
#define BUFDW   5120
#define BUFBYTES 20480

typedef __attribute__((ext_vector_type(16))) __bf16 v16bf;
typedef __attribute__((ext_vector_type(8)))  float  v8f;

union FragBF { v16bf v; unsigned u[8]; uint4 q[2]; };

static __device__ __forceinline__ unsigned short f2bf(float x) {
  return __builtin_bit_cast(unsigned short, __float2bfloat16(x));
}

// async Global->LDS DMA (gfx1250): per-lane 16B memory -> LDS, ASYNCcnt tracked
static __device__ __forceinline__ void async_ld_b128(unsigned lds_off,
                                                     unsigned long long gaddr) {
  asm volatile("global_load_async_to_lds_b128 %0, %1, off"
               :: "v"(lds_off), "v"(gaddr) : "memory");
}
static __device__ __forceinline__ void wait_async0() {
  asm volatile("s_wait_asynccnt 0x0" ::: "memory");
}
static __device__ __forceinline__ void wait_async5() {
  asm volatile("s_wait_asynccnt 0x5" ::: "memory");   // newest tile (5 DMAs) may stay in flight
}
static __device__ __forceinline__ unsigned lds_off32(const void* p) {
  return (unsigned)(unsigned long long)p;   // low 32 bits = LDS offset
}

// ---------------- threefry2x32 (JAX PRNG) ----------------
static __device__ __forceinline__ unsigned rotl32(unsigned x, unsigned n) {
  return (x << n) | (x >> (32u - n));
}
static __device__ void tf2x32(unsigned k0, unsigned k1, unsigned c0, unsigned c1,
                              unsigned& o0, unsigned& o1) {
  unsigned ks[3] = {k0, k1, k0 ^ k1 ^ 0x1BD11BDAu};
  unsigned x0 = c0 + ks[0], x1 = c1 + ks[1];
  const unsigned rA[4] = {13u, 15u, 26u, 6u};
  const unsigned rB[4] = {17u, 29u, 16u, 24u};
#pragma unroll
  for (int i = 0; i < 5; ++i) {
    const unsigned* r = (i & 1) ? rB : rA;
#pragma unroll
    for (int j = 0; j < 4; ++j) { x0 += x1; x1 = rotl32(x1, r[j]) ^ x0; }
    x0 += ks[(i + 1) % 3];
    x1 += ks[(i + 2) % 3] + (unsigned)(i + 1);
  }
  o0 = x0; o1 = x1;
}
static __device__ void derive_keys(unsigned& km0, unsigned& km1,
                                   unsigned& kn0, unsigned& kn1) {
  unsigned p00, p01, p10, p11;
  tf2x32(0u, 42u, 0u, 2u, p00, p01);
  tf2x32(0u, 42u, 1u, 3u, p10, p11);
  km0 = p00; km1 = p10;
  kn0 = p01; kn1 = p11;
}
static __device__ __forceinline__ float bits_to_unit(unsigned bits) {
  return __uint_as_float(0x3f800000u | (bits >> 9)) - 1.0f;
}
static __device__ __forceinline__ int label_len(int len) {
  return (((len - 3) / 2 + 1) - 3) / 2 + 1;
}

// ---------------- kernel 1: build padded bf16 "stacked" ----------------
__global__ void k_pack_stacked(const float* __restrict__ raw,
                               unsigned short* __restrict__ stackedB) {
  long long i = (long long)blockIdx.x * blockDim.x + threadIdx.x;
  int m  = (int)(i / KP);
  int kk = (int)(i % KP);
  float v = 0.0f;
  if (m < MROWS && kk < KIN) {
    int b  = m / T2_;
    int t2 = m % T2_;
    int d  = kk / 9;
    int r  = kk % 9;
    int k1 = r / 3;
    int k2 = r % 3;
    int tt = t2 * 4 + 2 * k2 + k1;
    v = raw[((long long)b * T_ + tt) * D_ + d];
  }
  stackedB[i] = f2bf(v);
}

// -------- kernel 2: projector -> fragment-major DMA chunks ---------------
// chunk (nb,kt) of 4096 dwords: chunk[nc*16+p] = {bf16 P[kt*32+2p][nb*256+nc],
//                                                 bf16 P[kt*32+2p+1][...]}
__global__ void k_pack_proj_pairs(const float* __restrict__ proj,
                                  unsigned* __restrict__ projP2) {
  int i = blockIdx.x * blockDim.x + threadIdx.x;   // < 2*NKT1*4096
  int chunk = i >> 12;
  int r = i & 4095;
  int nb = chunk / NKT1, kt = chunk % NKT1;
  int nc = r >> 4, p = r & 15;
  int k = kt * 32 + 2 * p;
  int n = nb * 256 + nc;
  float v0 = (k     < KIN) ? proj[(size_t)k * CBDIM + n]       : 0.0f;
  float v1 = (k + 1 < KIN) ? proj[(size_t)(k + 1) * CBDIM + n] : 0.0f;
  projP2[i] = (unsigned)f2bf(v0) | ((unsigned)f2bf(v1) << 16);
}

// ---------------- kernel 3: bf16 codebook + row norms ----------------
__global__ void k_pack_cb(const float* __restrict__ cb,
                          unsigned short* __restrict__ cbB,
                          float* __restrict__ cnorm) {
  int r = blockIdx.x, t = threadIdx.x;
  float s = 0.0f;
  for (int c = t; c < CBDIM; c += 256) {
    float v = cb[r * CBDIM + c];
    cbB[r * CBDIM + c] = f2bf(v);
    s += v * v;
  }
  __shared__ float red[256];
  red[t] = s; __syncthreads();
  for (int o = 128; o > 0; o >>= 1) {
    if (t < o) red[t] += red[t + o];
    __syncthreads();
  }
  if (t == 0) cnorm[r] = red[0];
}

// ---------------- kernel 4: GEMM1 targets = stacked @ proj (bf16 WMMA) ----
// block tile 64(M) x 256(N); 8 waves = 2(M) x 4(N); wave tile 32 x 64
// (8 WMMA per K-step per wave, each B frag reused by 2 A frags).
// Triple-buffered LDS, all-async staging, ONE barrier per K-step,
// rotating buffer offsets (no div/mod).
__global__ void __launch_bounds__(256)
k_gemm1(const unsigned short* __restrict__ stackedB,
        const unsigned* __restrict__ projP2,
        unsigned short* __restrict__ targetsB) {
  __shared__ __align__(16) unsigned lds[3 * BUFDW];

  const int tid  = threadIdx.x;
  const int lane = tid & 31;
  const int wid  = tid >> 5;
  const int waveM = wid & 1;
  const int waveN = wid >> 1;
  const int half  = lane >> 4;
  const int mrow  = lane & 15;
  const int bm0 = blockIdx.x * 64;
  const int nb  = blockIdx.y;            // N block (256 cols)

  const unsigned short* gA = stackedB + (size_t)(bm0 + (tid >> 2)) * KP + (tid & 3) * 8;
  const unsigned*       gB = projP2 + (size_t)nb * NKT1 * 4096 + tid * 16;

  const unsigned tA = lds_off32(lds) + tid * 16;          // A slot: 4 dwords
  const unsigned tB = lds_off32(lds) + 4096 + tid * 64;   // B slot: 16 dwords

#define STAGE1(j, off) do {                                                   \
    async_ld_b128(tA + (off), (unsigned long long)(gA + (j) * 32));           \
    const unsigned* _s = gB + (size_t)(j) * 4096;                             \
    async_ld_b128(tB + (off),      (unsigned long long)_s);                   \
    async_ld_b128(tB + (off) + 16, (unsigned long long)(_s + 4));             \
    async_ld_b128(tB + (off) + 32, (unsigned long long)(_s + 8));             \
    async_ld_b128(tB + (off) + 48, (unsigned long long)(_s + 12));            \
  } while (0)

  const v8f vzero = {0.f, 0.f, 0.f, 0.f, 0.f, 0.f, 0.f, 0.f};
  v8f acc[2][4] = {{vzero, vzero, vzero, vzero}, {vzero, vzero, vzero, vzero}};

  STAGE1(0, 0);
  STAGE1(1, BUFBYTES);
  unsigned rdOff = 0, stOff = 2 * BUFBYTES;

#pragma unroll 1
  for (int kk = 0; kk < NKT1; ++kk) {
    if (kk == NKT1 - 1) wait_async0(); else wait_async5();
    __syncthreads();
    const unsigned* bufA = lds + (rdOff >> 2);
    const unsigned* bufB = bufA + 1024;

    FragBF fa[2];
#pragma unroll
    for (int m2 = 0; m2 < 2; ++m2) {
      const unsigned* arow = bufA + (waveM * 32 + m2 * 16 + mrow) * 16 + 4 * half;
      fa[m2].q[0] = *(const uint4*)arow;
      fa[m2].q[1] = *(const uint4*)(arow + 8);
    }
#pragma unroll
    for (int s = 0; s < 4; ++s) {
      FragBF fb;
      const int nc = waveN * 64 + s * 16 + mrow;
      const uint4* bp = (const uint4*)&bufB[nc * 16 + 4 * half];
      fb.q[0] = bp[0];
      fb.q[1] = bp[2];
      acc[0][s] = __builtin_amdgcn_wmma_f32_16x16x32_bf16(
          false, fa[0].v, false, fb.v, (short)0, acc[0][s], false, false);
      acc[1][s] = __builtin_amdgcn_wmma_f32_16x16x32_bf16(
          false, fa[1].v, false, fb.v, (short)0, acc[1][s], false, false);
    }
    if (kk + 2 < NKT1) STAGE1(kk + 2, stOff);
    rdOff = (rdOff == 2 * BUFBYTES) ? 0u : rdOff + BUFBYTES;
    stOff = (stOff == 2 * BUFBYTES) ? 0u : stOff + BUFBYTES;
  }
#undef STAGE1

  // store D (lane = n + 16*(m/8); vgpr v -> m = v + 8*half)
#pragma unroll
  for (int m2 = 0; m2 < 2; ++m2) {
#pragma unroll
    for (int s = 0; s < 4; ++s) {
#pragma unroll
      for (int v = 0; v < 8; ++v) {
        int gr = bm0 + waveM * 32 + m2 * 16 + v + 8 * half;
        int gc = nb * 256 + waveN * 64 + s * 16 + mrow;
        targetsB[(size_t)gr * CBDIM + gc] = f2bf(acc[m2][s][v]);
      }
    }
  }
}

// ---------------- kernel 5: GEMM2 + fused argmin -> labels ----------------
// Flattened 4(nchunk of 256 cols) x 16(kstep) pipelined loop.
__global__ void __launch_bounds__(256)
k_gemm2_argmin(const unsigned short* __restrict__ targetsB,
               const unsigned short* __restrict__ cbB,
               const float* __restrict__ cnorm,
               const int* __restrict__ length,
               float* __restrict__ out_labels) {
  __shared__ __align__(16) unsigned lds[3 * BUFDW];
  __shared__ float rv[64 * 4];
  __shared__ int   ri[64 * 4];

  const int tid  = threadIdx.x;
  const int lane = tid & 31;
  const int wid  = tid >> 5;
  const int waveM = wid & 1;
  const int waveN = wid >> 1;
  const int half  = lane >> 4;
  const int mrow  = lane & 15;
  const int bm0 = blockIdx.x * 64;

  const unsigned short* gA = targetsB + (size_t)(bm0 + (tid >> 2)) * CBDIM + (tid & 3) * 8;
  const unsigned* cbU = (const unsigned*)cbB;   // 256 dwords per codebook row

  const unsigned tA = lds_off32(lds) + tid * 16;
  const unsigned tB = lds_off32(lds) + 4096 + tid * 64;

#define STAGE2(j, off) do {                                                   \
    int _kt = (j) & 15;                                                       \
    int _n0 = ((j) >> 4) * 256;                                               \
    async_ld_b128(tA + (off), (unsigned long long)(gA + _kt * 32));           \
    const unsigned* _s = cbU + (size_t)(_n0 + tid) * 256 + _kt * 16;          \
    async_ld_b128(tB + (off),      (unsigned long long)_s);                   \
    async_ld_b128(tB + (off) + 16, (unsigned long long)(_s + 4));             \
    async_ld_b128(tB + (off) + 32, (unsigned long long)(_s + 8));             \
    async_ld_b128(tB + (off) + 48, (unsigned long long)(_s + 12));            \
  } while (0)

  float rminv[2][8];
  int   rmini[2][8];
#pragma unroll
  for (int m2 = 0; m2 < 2; ++m2)
#pragma unroll
    for (int v = 0; v < 8; ++v) { rminv[m2][v] = 3.4e38f; rmini[m2][v] = 0; }

  const v8f vzero = {0.f, 0.f, 0.f, 0.f, 0.f, 0.f, 0.f, 0.f};
  v8f acc[2][4] = {{vzero, vzero, vzero, vzero}, {vzero, vzero, vzero, vzero}};

  STAGE2(0, 0);
  STAGE2(1, BUFBYTES);
  unsigned rdOff = 0, stOff = 2 * BUFBYTES;

#pragma unroll 1
  for (int kk = 0; kk < 64; ++kk) {
    if (kk == 63) wait_async0(); else wait_async5();
    __syncthreads();
    const unsigned* bufA = lds + (rdOff >> 2);
    const unsigned* bufB = bufA + 1024;

    FragBF fa[2];
#pragma unroll
    for (int m2 = 0; m2 < 2; ++m2) {
      const unsigned* arow = bufA + (waveM * 32 + m2 * 16 + mrow) * 16 + 4 * half;
      fa[m2].q[0] = *(const uint4*)arow;
      fa[m2].q[1] = *(const uint4*)(arow + 8);
    }
#pragma unroll
    for (int s = 0; s < 4; ++s) {
      FragBF fb;
      const int nc = waveN * 64 + s * 16 + mrow;
      const uint4* bp = (const uint4*)&bufB[nc * 16 + 4 * half];
      fb.q[0] = bp[0];
      fb.q[1] = bp[2];
      acc[0][s] = __builtin_amdgcn_wmma_f32_16x16x32_bf16(
          false, fa[0].v, false, fb.v, (short)0, acc[0][s], false, false);
      acc[1][s] = __builtin_amdgcn_wmma_f32_16x16x32_bf16(
          false, fa[1].v, false, fb.v, (short)0, acc[1][s], false, false);
    }
    if ((kk & 15) == 15) {
      // finished a full K sweep for this 256-column chunk: fold argmin
      const int n0 = (kk >> 4) * 256;
#pragma unroll
      for (int s = 0; s < 4; ++s) {
        const int gn = n0 + waveN * 64 + s * 16 + mrow;
        const float cn = cnorm[gn];
#pragma unroll
        for (int m2 = 0; m2 < 2; ++m2) {
#pragma unroll
          for (int v = 0; v < 8; ++v) {
            float d = cn - 2.0f * acc[m2][s][v];
            if (d < rminv[m2][v]) { rminv[m2][v] = d; rmini[m2][v] = gn; }
          }
          acc[m2][s] = vzero;
        }
      }
    }
    if (kk + 2 < 64) STAGE2(kk + 2, stOff);
    rdOff = (rdOff == 2 * BUFBYTES) ? 0u : rdOff + BUFBYTES;
    stOff = (stOff == 2 * BUFBYTES) ? 0u : stOff + BUFBYTES;
  }
#undef STAGE2

  // wave-level argmin reduction across the 16 lanes of each half-group
#pragma unroll
  for (int m2 = 0; m2 < 2; ++m2) {
#pragma unroll
    for (int v = 0; v < 8; ++v) {
      float bv = rminv[m2][v];
      int   bi = rmini[m2][v];
#pragma unroll
      for (int x = 1; x < 16; x <<= 1) {
        float ov = __shfl_xor(bv, x, 32);
        int   oi = __shfl_xor(bi, x, 32);
        if (ov < bv) { bv = ov; bi = oi; }
      }
      if (mrow == 0) {
        int r = waveM * 32 + m2 * 16 + v + 8 * half;
        rv[r * 4 + waveN] = bv;
        ri[r * 4 + waveN] = bi;
      }
    }
  }
  __syncthreads();
  if (tid < 64) {
    float best = rv[tid * 4];
    int   bi   = ri[tid * 4];
#pragma unroll
    for (int j = 1; j < 4; ++j) {
      float x = rv[tid * 4 + j];
      if (x < best) { best = x; bi = ri[tid * 4 + j]; }
    }
    int gm = bm0 + tid;
    if (gm < MROWS) {
      int b  = gm / T2_;
      int t2 = gm % T2_;
      int ll = label_len(length[b]);
      out_labels[gm] = (t2 < ll) ? (float)(bi + 1) : 0.0f;
    }
  }
}

// ---------------- kernel 6: mask RNG -> masked flags + masked_dim ----------
__global__ void k_rng_mask(const int* __restrict__ length,
                           unsigned char* __restrict__ maskf,
                           float* __restrict__ out_maskdim) {
  int i = blockIdx.x * blockDim.x + threadIdx.x;
  if (i >= MROWS) return;
  unsigned km0, km1, kn0, kn1;
  derive_keys(km0, km1, kn0, kn1);
  const int halfn = MROWS / 2;   // 8176
  unsigned c0 = (i < halfn) ? (unsigned)i : (unsigned)(i - halfn);
  unsigned c1 = c0 + (unsigned)halfn;
  unsigned y0, y1;
  tf2x32(km0, km1, c0, c1, y0, y1);
  unsigned bits = (i < halfn) ? y0 : y1;
  float u = bits_to_unit(bits);
  int b  = i / T2_;
  int t2 = i % T2_;
  int valid = (t2 < label_len(length[b])) ? 1 : 0;
  int m = (u < 0.1f) && valid;
  maskf[i] = (unsigned char)m;
  out_maskdim[i] = m ? 1.0f : 0.0f;
}

// ---------------- kernel 7: masked feats (noise where frame masked) -------
__global__ void k_masked_feats(const float* __restrict__ auged,
                               const unsigned char* __restrict__ maskf,
                               float* __restrict__ out_feats) {
  long long i = (long long)blockIdx.x * blockDim.x + threadIdx.x;
  if (i >= NFEAT) return;
  int b   = (int)(i / (T_ * D_));
  int rem = (int)(i % (T_ * D_));
  int t   = rem / D_;
  int t2a = t >> 2;
  int fm = 0;
  if (t2a < T2_ && maskf[b * T2_ + t2a]) fm = 1;
  if (!fm && t2a >= 1 && (t & 3) <= 2 && maskf[b * T2_ + t2a - 1]) fm = 1;
  float outv;
  if (fm) {
    unsigned km0, km1, kn0, kn1;
    derive_keys(km0, km1, kn0, kn1);
    const unsigned halfn = NFEAT / 2u;  // 2621440
    unsigned c0 = (i < (long long)halfn) ? (unsigned)i : (unsigned)(i - halfn);
    unsigned c1 = c0 + halfn;
    unsigned y0, y1;
    tf2x32(kn0, kn1, c0, c1, y0, y1);
    unsigned bits = (i < (long long)halfn) ? y0 : y1;
    float ub = bits_to_unit(bits);
    const float lo = -0.99999994f;      // nextafter(-1, 0)
    float u = lo + ub * (1.0f - lo);
    outv = 0.1f * 1.41421356237f * erfinvf(u);
  } else {
    outv = auged[i];
  }
  out_feats[i] = outv;
}

// ---------------- host launcher ----------------
extern "C" void kernel_launch(void* const* d_in, const int* in_sizes, int n_in,
                              void* d_out, int out_size, void* d_ws, size_t ws_size,
                              hipStream_t stream) {
  const float* raw    = (const float*)d_in[0];
  const float* auged  = (const float*)d_in[1];
  const int*   length = (const int*)d_in[2];
  const float* proj   = (const float*)d_in[3];
  const float* cb     = (const float*)d_in[4];

  char* ws = (char*)d_ws;
  size_t off = 0;
  unsigned short* stackedB = (unsigned short*)(ws + off); off += (size_t)MPAD * KP * 2;
  unsigned*       projP2   = (unsigned*)(ws + off);       off += (size_t)2 * NKT1 * 4096 * 4;
  unsigned short* targetsB = (unsigned short*)(ws + off); off += (size_t)MPAD * CBDIM * 2;
  unsigned short* cbB      = (unsigned short*)(ws + off); off += (size_t)CBSIZE * CBDIM * 2;
  float*          cnorm    = (float*)(ws + off);          off += 4096;
  unsigned char*  maskf    = (unsigned char*)(ws + off);  off += MROWS;

  float* out_feats   = (float*)d_out;
  float* out_labels  = out_feats + NFEAT;
  float* out_maskdim = out_labels + MROWS;

  // pack / convert
  k_pack_stacked<<<(MPAD * KP) / 256, 256, 0, stream>>>(raw, stackedB);
  k_pack_proj_pairs<<<(2 * NKT1 * 4096) / 256, 256, 0, stream>>>(proj, projP2);
  k_pack_cb<<<CBSIZE, 256, 0, stream>>>(cb, cbB, cnorm);

  // GEMM 1: targets
  k_gemm1<<<dim3(MPAD / 64, 2), 256, 0, stream>>>(stackedB, projP2, targetsB);

  // GEMM 2 + argmin: labels
  k_gemm2_argmin<<<MPAD / 64, 256, 0, stream>>>(targetsB, cbB, cnorm, length, out_labels);

  // RNG mask + masked_dim, then masked feats
  k_rng_mask<<<(MROWS + 255) / 256, 256, 0, stream>>>(length, maskf, out_maskdim);
  k_masked_feats<<<(NFEAT + 255) / 256, 256, 0, stream>>>(auged, maskf, out_feats);
}
